// RealNVP_22978075034123
// MI455X (gfx1250) — matrix operand order repaired and checked by reading, not compiled, not run on hardware
//
#include <hip/hip_runtime.h>
#include <hip/hip_bf16.h>

typedef __attribute__((ext_vector_type(16))) _Float16 v16h;
typedef __attribute__((ext_vector_type(8)))  _Float16 v8h;
typedef __attribute__((ext_vector_type(2)))  _Float16 v2h;
typedef __attribute__((ext_vector_type(8)))  float    v8f;

#define NB 32      // flow blocks
#define HH 64      // hidden dim
#define LIMIT 10.0f
#define CLAMPV 10000.0f

// ---------------------------------------------------------------------------
// Prep: pack W2 (32x64x64 f32) and W3 (32x64x2 f32, padded to 16 cols) into
// f16 WMMA B-operand layout:
//   B operand (32K x 16N): lane l -> N = l%16, h = l/16,
//   halves[j] = W[8h + (j&7) + 16*(j>>3)][N],  per K-chunk c (Koffset = 32c).
// B2 layout: [b][c][t][lane][16 halves]  (t = N-tile 0..3)
// B3 layout: [b][c][lane][16 halves]     (cols >= 2 are zero)
// ---------------------------------------------------------------------------
__global__ void realnvp_prep(const float* __restrict__ W2,
                             const float* __restrict__ W3,
                             _Float16* __restrict__ B2,
                             _Float16* __restrict__ B3) {
    int idx = blockIdx.x * blockDim.x + threadIdx.x;
    const int NB2 = NB * 2 * 4 * 32 * 16;       // 131072
    const int NB3 = NB * 2 * 32 * 16;           // 32768
    if (idx < NB2) {
        int j    = idx & 15;
        int lane = (idx >> 4) & 31;
        int t    = (idx >> 9) & 3;
        int c    = (idx >> 11) & 1;
        int b    = idx >> 12;
        int K    = 32 * c + 8 * (lane >> 4) + (j & 7) + 16 * (j >> 3);
        int n    = 16 * t + (lane & 15);
        B2[idx] = (_Float16)W2[b * 4096 + K * 64 + n];
    } else if (idx - NB2 < NB3) {
        int i2   = idx - NB2;
        int j    = i2 & 15;
        int lane = (i2 >> 4) & 31;
        int c    = (i2 >> 9) & 1;
        int b    = i2 >> 10;
        int K    = 32 * c + 8 * (lane >> 4) + (j & 7) + 16 * (j >> 3);
        int col  = lane & 15;
        B3[i2] = (col < 2) ? (_Float16)W3[b * 128 + K * 2 + col] : (_Float16)0.0f;
    }
}

// ---------------------------------------------------------------------------
// Main: one wave per 16-point tile, 8 waves (128 points) per workgroup.
// All 32 flow blocks processed in-register/LDS; WMMA for layers 2 and 3.
// ---------------------------------------------------------------------------
__global__ __launch_bounds__(256) void realnvp_main(
    const float* __restrict__ x,
    const float* __restrict__ W1, const float* __restrict__ b1,
    const float* __restrict__ b2, const float* __restrict__ b3,
    const float* __restrict__ loc, const float* __restrict__ lsc,
    const _Float16* __restrict__ B2g, const _Float16* __restrict__ B3g,
    float* __restrict__ partials, int N)
{
    __shared__ __align__(16) _Float16 hbuf[8][16][HH]; // per-wave h tile (f16)
    __shared__ float pbuf[8][16][2];                   // per-wave (shift,scale)
    __shared__ float wsum[8];

    const int w    = threadIdx.x >> 5;
    const int lane = threadIdx.x & 31;
    const int sub  = lane & 15;   // point row / N-col sub-index
    const int hi   = lane >> 4;

    const int base   = (blockIdx.x * 8 + w) * 16;
    const int ptIdx  = base + sub;
    const int ptClmp = (ptIdx < N) ? ptIdx : (N - 1);

    // hi-lanes mirror low lanes: no divergence in the scalar epilogue.
    float zx = x[ptClmp * 2 + 0];
    float zy = x[ptClmp * 2 + 1];
    float ld = 0.0f;
    bool  mask = true;

    const v16h* B2v = (const v16h*)B2g;
    const v16h* B3v = (const v16h*)B3g;

    for (int bi = NB - 1; bi >= 0; --bi) {
        // ---- step part 1: mask on pre-swap z, then swapped+clamped update ----
        bool in1 = (fabsf(zx) < LIMIT) && (fabsf(zy) < LIMIT);
        mask = mask && in1;
        if (mask) {
            float nx = fminf(fmaxf(zy, -CLAMPV), CLAMPV);
            float ny = fminf(fmaxf(zx, -CLAMPV), CLAMPV);
            zx = nx; zy = ny;
        }

        // ---- layer 1: h = relu(z1*W1 + b1); lane handles cols 2l, 2l+1 ----
        float2 w1p = ((const float2*)(W1 + bi * HH))[lane];
        float2 b1p = ((const float2*)(b1 + bi * HH))[lane];
#pragma unroll
        for (int p = 0; p < 16; ++p) {
            float z1p = __shfl(zx, p, 32);
            v2h hv;
            hv.x = (_Float16)fmaxf(z1p * w1p.x + b1p.x, 0.0f);
            hv.y = (_Float16)fmaxf(z1p * w1p.y + b1p.y, 0.0f);
            ((v2h*)&hbuf[w][p][0])[lane] = hv;
        }
        __syncthreads();

        // ---- build A operands (16x32 f16 layout, §7.12.2) from LDS ----
        const v8h* rowp = (const v8h*)&hbuf[w][sub][0]; // 8 units of 8 halves
        v8h lo0 = rowp[0 + hi], hp0 = rowp[2 + hi];     // K-chunk 0
        v8h lo1 = rowp[4 + hi], hp1 = rowp[6 + hi];     // K-chunk 1
        v16h a0 = __builtin_shufflevector(lo0, hp0, 0,1,2,3,4,5,6,7,8,9,10,11,12,13,14,15);
        v16h a1 = __builtin_shufflevector(lo1, hp1, 0,1,2,3,4,5,6,7,8,9,10,11,12,13,14,15);

        // ---- layer 2: 4 N-tiles x (K=64) WMMA, bias in C, relu, back to LDS ----
#pragma unroll
        for (int t = 0; t < 4; ++t) {
            float bias = b2[bi * HH + 16 * t + sub];
            v8f cc;
#pragma unroll
            for (int r = 0; r < 8; ++r) cc[r] = bias;
            v16h Bk0 = B2v[((bi * 2 + 0) * 4 + t) * 32 + lane];
            v16h Bk1 = B2v[((bi * 2 + 1) * 4 + t) * 32 + lane];
            cc = __builtin_amdgcn_wmma_f32_16x16x32_f16(false, a0, false, Bk0,
                                                        (short)0, cc, false, false);
            cc = __builtin_amdgcn_wmma_f32_16x16x32_f16(false, a1, false, Bk1,
                                                        (short)0, cc, false, false);
#pragma unroll
            for (int r = 0; r < 8; ++r)
                hbuf[w][r + 8 * hi][16 * t + sub] = (_Float16)fmaxf(cc[r], 0.0f);
        }
        __syncthreads();

        // ---- layer 3: param = h @ W3pad + b3 (one 16-col tile, cols 0/1 real) ----
        {
            const v8h* rp2 = (const v8h*)&hbuf[w][sub][0];
            v8h l0 = rp2[0 + hi], h0 = rp2[2 + hi];
            v8h l1 = rp2[4 + hi], h1 = rp2[6 + hi];
            v16h a20 = __builtin_shufflevector(l0, h0, 0,1,2,3,4,5,6,7,8,9,10,11,12,13,14,15);
            v16h a21 = __builtin_shufflevector(l1, h1, 0,1,2,3,4,5,6,7,8,9,10,11,12,13,14,15);
            float cv = (sub == 0) ? b3[bi * 2 + 0] : ((sub == 1) ? b3[bi * 2 + 1] : 0.0f);
            v8f c3;
#pragma unroll
            for (int r = 0; r < 8; ++r) c3[r] = cv;
            v16h Ck0 = B3v[(bi * 2 + 0) * 32 + lane];
            v16h Ck1 = B3v[(bi * 2 + 1) * 32 + lane];
            c3 = __builtin_amdgcn_wmma_f32_16x16x32_f16(false, a20, false, Ck0,
                                                        (short)0, c3, false, false);
            c3 = __builtin_amdgcn_wmma_f32_16x16x32_f16(false, a21, false, Ck1,
                                                        (short)0, c3, false, false);
            if (sub < 2) {
#pragma unroll
                for (int r = 0; r < 8; ++r) pbuf[w][r + 8 * hi][sub] = c3[r];
            }
        }
        __syncthreads();

        // ---- step part 2: inverse affine coupling + mask/clamp/logdet ----
        float shift = pbuf[w][sub][0];
        float scale = pbuf[w][sub][1];
        float z2n = (zy - shift) * expf(-scale);
        bool in2 = (fabsf(zx) < LIMIT) && (fabsf(zy) < LIMIT);
        mask = mask && in2;
        if (mask) {
            zx = fminf(fmaxf(zx,  -CLAMPV), CLAMPV);
            zy = fminf(fmaxf(z2n, -CLAMPV), CLAMPV);
            ld -= scale;
        }
    }

    // ---- base log-prob + deterministic reduction ----
    float loc0 = loc[0], loc1 = loc[1];
    float ls0 = lsc[0], ls1 = lsc[1];
    float t0 = (zx - loc0) * expf(-ls0);
    float t1 = (zy - loc1) * expf(-ls1);
    float lp = -1.8378770664093453f - (ls0 + 0.5f * t0 * t0) - (ls1 + 0.5f * t1 * t1);
    float val = (hi == 0 && ptIdx < N) ? (lp + ld) : 0.0f;
#pragma unroll
    for (int off = 16; off >= 1; off >>= 1) val += __shfl_xor(val, off, 32);
    if (lane == 0) wsum[w] = val;
    __syncthreads();
    if (threadIdx.x == 0) {
        float s = 0.0f;
        for (int i = 0; i < 8; ++i) s += wsum[i];   // fixed order: deterministic
        partials[blockIdx.x] = s;
    }
}

__global__ __launch_bounds__(256) void realnvp_reduce(const float* __restrict__ partials,
                                                      int n, float* __restrict__ out) {
    __shared__ float red[256];
    float s = 0.0f;
    for (int i = threadIdx.x; i < n; i += 256) s += partials[i]; // fixed stride order
    red[threadIdx.x] = s;
    __syncthreads();
    if (threadIdx.x == 0) {
        float tot = 0.0f;
        for (int i = 0; i < 256; ++i) tot += red[i];             // fixed order
        out[0] = tot;
    }
}

extern "C" void kernel_launch(void* const* d_in, const int* in_sizes, int n_in,
                              void* d_out, int out_size, void* d_ws, size_t ws_size,
                              hipStream_t stream) {
    const float* x   = (const float*)d_in[0];
    const float* W1  = (const float*)d_in[1];
    const float* b1v = (const float*)d_in[2];
    const float* W2  = (const float*)d_in[3];
    const float* b2v = (const float*)d_in[4];
    const float* W3  = (const float*)d_in[5];
    const float* b3v = (const float*)d_in[6];
    const float* loc = (const float*)d_in[7];
    const float* lsc = (const float*)d_in[8];

    const int N = in_sizes[0] / 2;

    // d_ws layout: B2 (256 KB f16) | B3 (64 KB f16) | partials (grid floats)
    _Float16* B2 = (_Float16*)d_ws;
    _Float16* B3 = (_Float16*)((char*)d_ws + 262144);
    float* partials = (float*)((char*)d_ws + 262144 + 65536);

    // 1) pack weights into WMMA B-operand f16 layout
    const int prepElems = NB * 2 * 4 * 32 * 16 + NB * 2 * 32 * 16; // 163840
    realnvp_prep<<<(prepElems + 255) / 256, 256, 0, stream>>>(W2, W3, B2, B3);

    // 2) flow inverse pass (128 points per workgroup)
    const int grid = (N + 127) / 128;
    realnvp_main<<<grid, 256, 0, stream>>>(x, W1, b1v, b2v, b3v, loc, lsc,
                                           B2, B3, partials, N);

    // 3) deterministic final reduction -> d_out[0]
    realnvp_reduce<<<1, 256, 0, stream>>>(partials, grid, (float*)d_out);
}
